// SNN_80985903333813
// MI455X (gfx1250) — matrix operand order repaired and checked by reading, not compiled, not run on hardware
//
#include <hip/hip_runtime.h>

// ---------------------------------------------------------------------------
// 3-layer LIF SNN on MI455X (gfx1250). B=256, T=64, I=512, H=1024, O=2.
// 3 kernels total: each layer runs its full T=64 recurrence in-kernel with
// membrane state in VGPRs; layer L reads layer L-1's f32 spike record from
// d_out (exact 0/1 -> f16). WMMA f16 with f32 accum; LIF folded into the
// accumulator init (C = 0.5*mem - reset).
// CDNA5 specifics: weight slab (64xK f16) staged ONCE per block into 320KB
// LDS via global_load_async_to_lds_b128 + s_wait_asynccnt; full A tile staged
// per timestep so the K loop is barrier-free ds_load_b128 + v_wmma; record
// stores use compile-time N for constant-offset addressing.
// ---------------------------------------------------------------------------

typedef __attribute__((ext_vector_type(16))) _Float16 v16h;
typedef __attribute__((ext_vector_type(8)))  _Float16 v8h;
typedef __attribute__((ext_vector_type(8)))  float    v8f;

#define SNN_B   256
#define SNN_T   64
#define SNN_I   512
#define SNN_H   1024
#define SNN_O   2

union Frag16 { v16h v; struct { v8h lo; v8h hi; } p; };

// A fragment (16x32 f16), k order used here (B uses the same):
// lane L (row m=L&15): elems 0..15 = K[(L>>4)*16 .. +15] (contiguous)
__device__ __forceinline__ v16h load_a_frag(const _Float16* As, int bstr,
                                            int row, int lane, int k0) {
    const int hi = lane >> 4;
    const _Float16* base = As + (size_t)row * bstr + k0;
    Frag16 u;
    u.p.lo = *(const v8h*)(base + hi * 16);
    u.p.hi = *(const v8h*)(base + hi * 16 + 8);
    return u.v;
}

// B fragment (32x16 f16), k order used here (same as A):
// lane L (col n=L&15): elems 0..15 = K[(L>>4)*16 .. +15] (contiguous)
__device__ __forceinline__ v16h load_b_frag(const _Float16* Bs, int bstr,
                                            int row, int lane, int k0) {
    const _Float16* base = Bs + (size_t)row * bstr + k0 + (lane >> 4) * 16;
    Frag16 u;
    u.p.lo = *(const v8h*)(base);
    u.p.hi = *(const v8h*)(base + 8);
    return u.v;
}

__device__ __forceinline__ v8h cvt8(float4 a, float4 b) {
    v8h h;
    h[0] = (_Float16)a.x; h[1] = (_Float16)a.y;
    h[2] = (_Float16)a.z; h[3] = (_Float16)a.w;
    h[4] = (_Float16)b.x; h[5] = (_Float16)b.y;
    h[6] = (_Float16)b.z; h[7] = (_Float16)b.w;
    return h;
}

// ---------------------------------------------------------------------------
// Fused all-timesteps GEMM + LIF, layers 0/1.  cur_t = A_t @ W^T, W f16 [N,K].
// Block: 128 thr = 4 waves, tile 64x64(=N slice), wave 32x32. grid=(B/64,N/64).
// LDS: Bs[64][K+8] f16 (whole weight slab, async DMA, loaded once)
//      As[64][K+8] f16 (whole A tile, restaged per timestep, f32->f16)
// Inner K loop: ds_load + wmma only (no barriers).
// ---------------------------------------------------------------------------
template<int K, int LDA, long TSTR>
__global__ __launch_bounds__(128)
void lif_layer_kernel(const float* __restrict__ A,
                      const _Float16* __restrict__ W,
                      float* __restrict__ memRec, float* __restrict__ spkRec,
                      const float* __restrict__ thrP) {
    constexpr int N    = SNN_H;     // both big layers have N = 1024
    constexpr int BSTR = K + 8;     // padded row stride (halves); 2064B/2048B+16
    extern __shared__ char smem[];
    _Float16* Bs = (_Float16*)smem;                          // 64*BSTR halves
    _Float16* As = (_Float16*)(smem + (size_t)64 * BSTR * 2);

    const int tid  = threadIdx.x;
    const int lane = tid & 31;
    const int wv   = tid >> 5;
    const int wm   = (wv >> 1) * 32;
    const int wn   = (wv & 1) * 32;
    const float thr = *thrP;

    const int srow  = tid >> 1;          // staging: 2 threads per row
    const int shalf = (tid & 1) * (K / 2);

    // ---- one-time: 64xK f16 weight slab -> LDS via async DMA ---------------
    {
        const unsigned ldsRowB =
            (unsigned)(size_t)(Bs + (size_t)srow * BSTR + shalf);
        const char* gRowB =
            (const char*)(W + (size_t)(blockIdx.y * 64 + srow) * K + shalf);
        for (int c = 0; c < K; c += 16) {  // K bytes per thread, 16B chunks
            asm volatile("global_load_async_to_lds_b128 %0, %1, off"
                         :: "v"(ldsRowB + c), "v"(gRowB + c) : "memory");
        }
    }
    asm volatile("s_wait_asynccnt 0x0" ::: "memory");
    __syncthreads();

    // ---- persistent membrane state in VGPRs --------------------------------
    float mem[2][2][8];
    #pragma unroll
    for (int i = 0; i < 2; ++i)
        #pragma unroll
        for (int j = 0; j < 2; ++j)
            #pragma unroll
            for (int e = 0; e < 8; ++e) mem[i][j][e] = 0.0f;

    const size_t aRowOff = (size_t)(blockIdx.x * 64 + srow) * (size_t)LDA + shalf;
    _Float16* asRow = As + (size_t)srow * BSTR + shalf;
    const int hi8  = (lane >> 4) * 8;
    const int col0 = lane & 15;

    for (int t = 0; t < SNN_T; ++t) {
        // LIF fold: C_init = 0.5*mem_prev - (mem_prev > thr)*thr
        v8f acc[2][2];
        #pragma unroll
        for (int i = 0; i < 2; ++i)
            #pragma unroll
            for (int j = 0; j < 2; ++j)
                #pragma unroll
                for (int e = 0; e < 8; ++e) {
                    const float mp = mem[i][j][e];
                    acc[i][j][e] = 0.5f * mp - ((mp > thr) ? thr : 0.0f);
                }

        // ---- stage full A tile for this timestep (f32 -> f16) -------------
        const float* gA = A + (long)t * TSTR + aRowOff;
        #pragma unroll 4
        for (int c = 0; c < K / 2; c += 16) {
            const float4 fa = *(const float4*)(gA + c);
            const float4 fb = *(const float4*)(gA + c + 4);
            const float4 fc = *(const float4*)(gA + c + 8);
            const float4 fd = *(const float4*)(gA + c + 12);
            *(v8h*)(asRow + c)     = cvt8(fa, fb);
            *(v8h*)(asRow + c + 8) = cvt8(fc, fd);
        }
        if (t + 1 < SNN_T) __builtin_prefetch(gA + TSTR, 0, 1);  // uniform
        __syncthreads();

        // ---- barrier-free K loop: ds_load_b128 + v_wmma -------------------
        for (int k0 = 0; k0 < K; k0 += 32) {
            const v16h a0 = load_a_frag(As, BSTR, wm + col0, lane, k0);
            const v16h a1 = load_a_frag(As, BSTR, wm + 16 + col0, lane, k0);
            const v16h b0 = load_b_frag(Bs, BSTR, wn + col0, lane, k0);
            const v16h b1 = load_b_frag(Bs, BSTR, wn + 16 + col0, lane, k0);

            acc[0][0] = __builtin_amdgcn_wmma_f32_16x16x32_f16(false, a0, false, b0,
                            (short)0, acc[0][0], false, false);
            acc[0][1] = __builtin_amdgcn_wmma_f32_16x16x32_f16(false, a0, false, b1,
                            (short)0, acc[0][1], false, false);
            acc[1][0] = __builtin_amdgcn_wmma_f32_16x16x32_f16(false, a1, false, b0,
                            (short)0, acc[1][0], false, false);
            acc[1][1] = __builtin_amdgcn_wmma_f32_16x16x32_f16(false, a1, false, b1,
                            (short)0, acc[1][1], false, false);
        }

        // ---- records: base pointer per tile, constant offsets r*N ---------
        float* mR = memRec + (size_t)t * SNN_B * N;
        float* sR = spkRec + (size_t)t * SNN_B * N;
        #pragma unroll
        for (int i = 0; i < 2; ++i) {
            #pragma unroll
            for (int j = 0; j < 2; ++j) {
                const int gb = blockIdx.x * 64 + wm + i * 16 + hi8;
                const int gn = blockIdx.y * 64 + wn + j * 16 + col0;
                float* pm = mR + (size_t)gb * N + gn;
                float* ps = sR + (size_t)gb * N + gn;
                #pragma unroll
                for (int e = 0; e < 8; ++e) {
                    const float m = acc[i][j][e];
                    mem[i][j][e] = m;
                    pm[e * N] = m;                               // offset:e*4096
                    ps[e * N] = ((m - thr) > 0.0f) ? 1.0f : 0.0f;
                }
            }
        }
        __syncthreads();   // As may be overwritten next t
    }
}

// ---------------------------------------------------------------------------
// Layer 2 head: [B,2] = spk1_t @ W2^T, t-loop in-kernel, one block per b.
// ---------------------------------------------------------------------------
__global__ __launch_bounds__(128)
void lif_out_kernel(const float* __restrict__ spk1Rec,
                    const float* __restrict__ W2,
                    float* __restrict__ memRec, float* __restrict__ spkRec,
                    const float* __restrict__ thrP) {
    __shared__ float red0[128];
    __shared__ float red1[128];
    const int b = blockIdx.x, tid = threadIdx.x;
    const float thr = *thrP;
    float memv = 0.0f;                     // meaningful in tid 0/1 only
    for (int t = 0; t < SNN_T; ++t) {
        const float* row = spk1Rec + (size_t)t * SNN_B * SNN_H + (size_t)b * SNN_H;
        float s0 = 0.f, s1 = 0.f;
        for (int i = tid; i < SNN_H; i += 128) {
            const float v = row[i];
            s0 += v * W2[i];
            s1 += v * W2[SNN_H + i];
        }
        red0[tid] = s0; red1[tid] = s1;
        __syncthreads();
        for (int off = 64; off > 0; off >>= 1) {
            if (tid < off) { red0[tid] += red0[tid + off]; red1[tid] += red1[tid + off]; }
            __syncthreads();
        }
        if (tid < 2) {
            const float cur   = (tid == 0) ? red0[0] : red1[0];
            const float reset = (memv > thr) ? thr : 0.0f;
            const float m     = 0.5f * memv + cur - reset;
            memv = m;
            const size_t idx = (size_t)t * SNN_B * SNN_O + b * SNN_O + tid;
            memRec[idx] = m;
            spkRec[idx] = ((m - thr) > 0.0f) ? 1.0f : 0.0f;
        }
        __syncthreads();
    }
}

__global__ void cvt_f32_f16_kernel(const float* __restrict__ s,
                                   _Float16* __restrict__ d, int n) {
    int i = blockIdx.x * blockDim.x + threadIdx.x;
    if (i < n) d[i] = (_Float16)s[i];
}

// ---------------------------------------------------------------------------
extern "C" void kernel_launch(void* const* d_in, const int* in_sizes, int n_in,
                              void* d_out, int out_size, void* d_ws, size_t ws_size,
                              hipStream_t stream) {
    (void)in_sizes; (void)n_in; (void)out_size; (void)ws_size;

    const float* x    = (const float*)d_in[0];   // [B,T,I]
    const float* W0   = (const float*)d_in[1];   // [H,I]
    const float* W1   = (const float*)d_in[2];   // [H,H]
    const float* W2   = (const float*)d_in[3];   // [O,H]
    const float* thr0 = (const float*)d_in[4];
    const float* thr1 = (const float*)d_in[5];
    const float* thr2 = (const float*)d_in[6];
    // d_in[7] = num_steps (== T = 64, hardcoded)

    // output sections: spk0 spk1 spk2 mem0 mem1 mem2
    const size_t SZH = (size_t)SNN_T * SNN_B * SNN_H;   // 16,777,216
    const size_t SZO = (size_t)SNN_T * SNN_B * SNN_O;   //     32,768
    float* out     = (float*)d_out;
    float* spk0Rec = out;
    float* spk1Rec = out + SZH;
    float* spk2Rec = out + 2 * SZH;
    float* mem0Rec = out + 2 * SZH + SZO;
    float* mem1Rec = out + 3 * SZH + SZO;
    float* mem2Rec = out + 4 * SZH + SZO;

    // workspace: two f16 weight slabs (3 MB)
    char* ws = (char*)d_ws;
    _Float16* W0h = (_Float16*)ws;  ws += (size_t)SNN_H * SNN_I * 2;
    _Float16* W1h = (_Float16*)ws;  ws += (size_t)SNN_H * SNN_H * 2;

    const int nW0 = SNN_H * SNN_I;
    const int nW1 = SNN_H * SNN_H;
    cvt_f32_f16_kernel<<<(nW0 + 255) / 256, 256, 0, stream>>>(W0, W0h, nW0);
    cvt_f32_f16_kernel<<<(nW1 + 255) / 256, 256, 0, stream>>>(W1, W1h, nW1);

    // dynamic LDS: Bs + As = 2 * 64*(K+8)*2 bytes
    const size_t shm0 = 2 * (size_t)64 * (SNN_I + 8) * 2;  // 133,120 B
    const size_t shm1 = 2 * (size_t)64 * (SNN_H + 8) * 2;  // 264,192 B (<320K/WGP)

    auto k0fn = lif_layer_kernel<SNN_I, SNN_T * SNN_I, (long)SNN_I>;
    auto k1fn = lif_layer_kernel<SNN_H, SNN_H, (long)SNN_B * SNN_H>;
    hipFuncSetAttribute((const void*)k0fn,
                        hipFuncAttributeMaxDynamicSharedMemorySize, (int)shm0);
    hipFuncSetAttribute((const void*)k1fn,
                        hipFuncAttributeMaxDynamicSharedMemorySize, (int)shm1);

    const dim3 gridG(SNN_B / 64, SNN_H / 64);   // (4, 16)

    // layer 0: A = x (per-t offset t*I, row stride T*I), K = I
    k0fn<<<gridG, 128, shm0, stream>>>(x, W0h, mem0Rec, spk0Rec, thr0);
    // layer 1: A = spk0 record (per-t offset B*H, row stride H), K = H
    k1fn<<<gridG, 128, shm1, stream>>>(spk0Rec, W1h, mem1Rec, spk1Rec, thr1);
    // layer 2 head
    lif_out_kernel<<<SNN_B, 128, 0, stream>>>(spk1Rec, W2, mem2Rec, spk2Rec, thr2);
}
